// MSNEAutoEncoder_78589311582741
// MI455X (gfx1250) — matrix operand order repaired and checked by reading, not compiled, not run on hardware
//
#include <hip/hip_runtime.h>
#include <hip/hip_bf16.h>

#define N_NODES 8192
#define HID 32

typedef __attribute__((ext_vector_type(16))) __bf16 v16bf;
typedef __attribute__((ext_vector_type(8)))  float  v8f;

static __device__ __forceinline__ unsigned int pack_bf16x2_rne(float lo, float hi) {
    unsigned int a = __float_as_uint(lo); a += 0x7FFFu + ((a >> 16) & 1u);
    unsigned int b = __float_as_uint(hi); b += 0x7FFFu + ((b >> 16) & 1u);
    return (a >> 16) | (b & 0xFFFF0000u);
}

static __device__ __forceinline__ unsigned short f32_to_bf16_rne(float f) {
    unsigned int u = __float_as_uint(f);
    u += 0x7FFFu + ((u >> 16) & 1u);
    return (unsigned short)(u >> 16);
}

// ---------------------------------------------------------------------------
// fp32 -> bf16 conversion (weights; done once, tiny)
// ---------------------------------------------------------------------------
__global__ void cvt_f32_bf16(const float* __restrict__ src,
                             unsigned short* __restrict__ dst, int n) {
    int i = blockIdx.x * blockDim.x + threadIdx.x;
    if (i < n) dst[i] = f32_to_bf16_rne(src[i]);
}

// ---------------------------------------------------------------------------
// Tiled WMMA GEMM:  C = relu(A[M,K](f32) * Bw[N,K](bf16)^T + bias[N])
// Block = 256 threads = 8 waves. Block tile: 32 (M) x 256 (N).
// Each wave: 2 M-tiles x 2 N-tiles = 32x32 output, 4 f32 accumulators.
// K-round = 64: stage 32x64 fp32 A-tile into LDS as bf16 (4 KB), then
// 2 sub-steps of K=32 -> 8 v_wmma per barrier pair.
// grid = (M/32, N/256). K must be a multiple of 64.
// ---------------------------------------------------------------------------
__global__ __launch_bounds__(256)
void gemm_bf16_relu(const float* __restrict__ A,
                    const unsigned short* __restrict__ Bw,
                    const float* __restrict__ bias,
                    float* __restrict__ C,
                    int M, int N, int K) {
    __shared__ __align__(16) unsigned short ldsA[32 * 64];   // 4 KB bf16 tile

    const int tid     = threadIdx.x;
    const int lane    = tid & 31;
    const int wv      = tid >> 5;
    const int m0      = blockIdx.x * 32;
    const int nbase   = blockIdx.y * 256 + wv * 32;   // wave's 32-col span
    const int nl      = lane & 15;                    // N / A-row local index
    const int halfsel = lane >> 4;                    // 0: lanes 0-15, 1: 16-31
    const int kbA     = halfsel * 16;                 // A fragment K base
    const int kbB     = halfsel * 16;                 // B fragment K base

    // B row pointers for the wave's two N-tiles (K-contiguous rows)
    const unsigned short* bp0 = Bw + (size_t)(nbase + nl) * K + kbB;
    const unsigned short* bp1 = bp0 + (size_t)16 * K;

    v8f c00 = {}, c01 = {}, c10 = {}, c11 = {};   // c[mt][nt]

    for (int k0 = 0; k0 < K; k0 += 64) {
        __syncthreads();
        // Stage 32x64 fp32 -> bf16: 512 float4 loads, 2 per thread, coalesced
        #pragma unroll
        for (int r = 0; r < 2; ++r) {
            int f    = tid + r * 256;            // float4 index
            int row  = f >> 4;                   // 16 float4 per 64-wide row
            int col4 = (f & 15) * 4;
            const float4 v = *(const float4*)(A + (size_t)(m0 + row) * K + k0 + col4);
            uint2 p;
            p.x = pack_bf16x2_rne(v.x, v.y);
            p.y = pack_bf16x2_rne(v.z, v.w);
            *(uint2*)(&ldsA[row * 64 + col4]) = p;
        }
        __syncthreads();

        #pragma unroll
        for (int ks = 0; ks < 64; ks += 32) {
            union Frag { uint4 q[2]; v16bf v; } a0, a1, b0, b1;
            // A fragments for the two M-tiles (contiguous K: element s = k 16*halfsel + s)
            const unsigned short* r0 = &ldsA[(0 * 16 + nl) * 64 + ks];
            const unsigned short* r1 = &ldsA[(1 * 16 + nl) * 64 + ks];
            a0.q[0] = *(const uint4*)(r0 + kbA);
            a0.q[1] = *(const uint4*)(r0 + kbA + 8);
            a1.q[0] = *(const uint4*)(r1 + kbA);
            a1.q[1] = *(const uint4*)(r1 + kbA + 8);
            // B fragments: 16 contiguous bf16 along K
            const unsigned short* p0 = bp0 + k0 + ks;
            const unsigned short* p1 = bp1 + k0 + ks;
            b0.q[0] = *(const uint4*)(p0);
            b0.q[1] = *(const uint4*)(p0 + 8);
            b1.q[0] = *(const uint4*)(p1);
            b1.q[1] = *(const uint4*)(p1 + 8);

            c00 = __builtin_amdgcn_wmma_f32_16x16x32_bf16(false, a0.v, false, b0.v,
                                                          (short)0, c00, false, false);
            c01 = __builtin_amdgcn_wmma_f32_16x16x32_bf16(false, a0.v, false, b1.v,
                                                          (short)0, c01, false, false);
            c10 = __builtin_amdgcn_wmma_f32_16x16x32_bf16(false, a1.v, false, b0.v,
                                                          (short)0, c10, false, false);
            c11 = __builtin_amdgcn_wmma_f32_16x16x32_bf16(false, a1.v, false, b1.v,
                                                          (short)0, c11, false, false);
        }
    }

    // Epilogue: C layout -> VGPR v: M = v + 8*halfsel, N = lane&15
    const int n0 = nbase + nl;
    const int n1 = n0 + 16;
    const float bv0 = bias[n0];
    const float bv1 = bias[n1];
    #pragma unroll
    for (int mt = 0; mt < 2; ++mt) {
        const v8f& cA = mt ? c10 : c00;
        const v8f& cB = mt ? c11 : c01;
        const int mbase = m0 + mt * 16 + halfsel * 8;
        #pragma unroll
        for (int v = 0; v < 8; ++v) {
            size_t row = (size_t)(mbase + v) * N;
            C[row + n0] = fmaxf(cA[v] + bv0, 0.0f);
            C[row + n1] = fmaxf(cB[v] + bv1, 0.0f);
        }
    }
}

// ---------------------------------------------------------------------------
// Small dense layer, fp32 VALU: C[m,n] = relu(dot(A[m,:], W[n,:]) + b[n])
// ---------------------------------------------------------------------------
__global__ void dense_relu_f32(const float* __restrict__ A,
                               const float* __restrict__ W,
                               const float* __restrict__ b,
                               float* __restrict__ C,
                               int M, int N, int K) {
    int idx = blockIdx.x * blockDim.x + threadIdx.x;
    if (idx >= M * N) return;
    int m = idx / N, n = idx - m * N;
    const float* a = A + (size_t)m * K;
    const float* w = W + (size_t)n * K;
    float acc = b[n];
    for (int k = 0; k < K; ++k) acc = fmaf(a[k], w[k], acc);
    C[idx] = fmaxf(acc, 0.0f);
}

// ---------------------------------------------------------------------------
// Z[i,d] = H[i,d] + sum_k Q[i, topk[i,k]] * H[topk[i,k], d]
// One wave per node; lane = feature dim (HID == 32 == wave32)
// ---------------------------------------------------------------------------
__global__ void qstar_f32(const float* __restrict__ Q,
                          const int* __restrict__ topk,
                          const float* __restrict__ H,
                          float* __restrict__ Z, int n_nodes) {
    int node = blockIdx.x * (blockDim.x >> 5) + (threadIdx.x >> 5);
    int d = threadIdx.x & 31;
    if (node >= n_nodes) return;
    const int*   tk   = topk + (size_t)node * 32;
    const float* qrow = Q + (size_t)node * n_nodes;
    float acc = H[(size_t)node * HID + d];
    for (int k = 0; k < 32; ++k) {
        int j = tk[k];
        acc = fmaf(qrow[j], H[(size_t)j * HID + d], acc);
    }
    Z[(size_t)node * HID + d] = acc;
}

// ---------------------------------------------------------------------------
extern "C" void kernel_launch(void* const* d_in, const int* in_sizes, int n_in,
                              void* d_out, int out_size, void* d_ws, size_t ws_size,
                              hipStream_t stream) {
    const float* X    = (const float*)d_in[0];
    const float* Q    = (const float*)d_in[1];
    const int*   topk = (const int*)  d_in[2];
    const float* We1  = (const float*)d_in[3];
    const float* be1  = (const float*)d_in[4];
    const float* We2  = (const float*)d_in[5];
    const float* be2  = (const float*)d_in[6];
    const float* We3  = (const float*)d_in[7];
    const float* be3  = (const float*)d_in[8];
    const float* Wd1  = (const float*)d_in[9];
    const float* bd1  = (const float*)d_in[10];
    const float* Wd2  = (const float*)d_in[11];
    const float* bd2  = (const float*)d_in[12];

    const int N = N_NODES;

    float* Xrec = (float*)d_out;                       // [N, N]
    float* Z    = (float*)d_out + (size_t)N * N;       // [N, HID]

    // Workspace layout (256B aligned chunks), total ~29 MB
    char* ws = (char*)d_ws;
    size_t off = 0;
    auto alloc = [&](size_t bytes) { char* p = ws + off; off = (off + bytes + 255) & ~(size_t)255; return p; };
    unsigned short* We1b = (unsigned short*)alloc((size_t)256 * N * 2);   // [256, 8192] bf16
    unsigned short* Wd2b = (unsigned short*)alloc((size_t)N * 256 * 2);   // [8192, 256] bf16
    float* H1 = (float*)alloc((size_t)N * 256 * 4);                       // [8192, 256]
    float* H2 = (float*)alloc((size_t)N * 84  * 4);                       // [8192, 84]
    float* Hf = (float*)alloc((size_t)N * HID * 4);                       // [8192, 32]
    float* Hd = (float*)alloc((size_t)N * 256 * 4);                       // [8192, 256]
    (void)ws_size; (void)in_sizes; (void)n_in; (void)out_size;

    // 0) weight conversions f32 -> bf16
    {
        int n = 256 * N;
        cvt_f32_bf16<<<(n + 255) / 256, 256, 0, stream>>>(We1, We1b, n);
        cvt_f32_bf16<<<(n + 255) / 256, 256, 0, stream>>>(Wd2, Wd2b, n);
    }

    // 1) H1 = relu(X @ We1^T + be1)   M=8192, N=256, K=8192
    gemm_bf16_relu<<<dim3(N / 32, 256 / 256), 256, 0, stream>>>(
        X, We1b, be1, H1, N, 256, N);

    // 2) H2 = relu(H1 @ We2^T + be2)  K=256, N=84
    {
        int total = N * 84;
        dense_relu_f32<<<(total + 255) / 256, 256, 0, stream>>>(
            H1, We2, be2, H2, N, 84, 256);
    }

    // 3) H = relu(H2 @ We3^T + be3)   K=84, N=32
    {
        int total = N * HID;
        dense_relu_f32<<<(total + 255) / 256, 256, 0, stream>>>(
            H2, We3, be3, Hf, N, HID, 84);
    }

    // 4) Z = H + Q_star (weighted neighbor gather) -> d_out tail
    qstar_f32<<<N / 8, 256, 0, stream>>>(Q, topk, Hf, Z, N);

    // 5) Hd = relu(Z @ Wd1^T + bd1)   K=32, N=256
    {
        int total = N * 256;
        dense_relu_f32<<<(total + 255) / 256, 256, 0, stream>>>(
            Z, Wd1, bd1, Hd, N, 256, HID);
    }

    // 6) X_rec = relu(Hd @ Wd2^T + bd2)  M=8192, N=8192, K=256 -> d_out head
    gemm_bf16_relu<<<dim3(N / 32, N / 256), 256, 0, stream>>>(
        Hd, Wd2b, bd2, Xrec, N, N, 256);
}
